// ImprovedGNNEncoder_14388140441813
// MI455X (gfx1250) — matrix operand order repaired and checked by reading, not compiled, not run on hardware
//
#include <hip/hip_runtime.h>
#include <hip/hip_bf16.h>

typedef __attribute__((ext_vector_type(16))) _Float16 v16h;
typedef __attribute__((ext_vector_type(8)))  _Float16 v8h;
typedef __attribute__((ext_vector_type(2)))  _Float16 v2h;
typedef __attribute__((ext_vector_type(8)))  float    v8f;

#define IN_C   128
#define HID_C  128
#define OUT_C  64
#define BN_EPS 1e-5f

// ---------------------------------------------------------------------------
// small utility kernels
// ---------------------------------------------------------------------------
__global__ void k_fill(float* __restrict__ p, float v, int n) {
  int i = blockIdx.x * blockDim.x + threadIdx.x;
  if (i < n) p[i] = v;
}

__global__ void k_deg_accum(const long long* __restrict__ dst, float* __restrict__ deg, int E) {
  int e = blockIdx.x * blockDim.x + threadIdx.x;
  if (e < E) atomicAdd(&deg[(size_t)dst[e]], 1.0f);
}

__global__ void k_rsqrt_inplace(float* __restrict__ p, int n) {
  int i = blockIdx.x * blockDim.x + threadIdx.x;
  if (i < n) {
    float d = p[i];
    p[i] = (d > 0.0f) ? rsqrtf(d) : 0.0f;
  }
}

__global__ void k_norm(const long long* __restrict__ src, const long long* __restrict__ dst,
                       const float* __restrict__ dinv, float* __restrict__ nrm, int E) {
  int e = blockIdx.x * blockDim.x + threadIdx.x;
  if (e < E) nrm[e] = dinv[(size_t)src[e]] * dinv[(size_t)dst[e]];
}

// pairwise f32 -> f16 conversion (n must be even): emits v_cvt_pk_f16_f32
__global__ void k_f32_to_f16(const float* __restrict__ src, _Float16* __restrict__ dst, int n2) {
  int i = blockIdx.x * blockDim.x + threadIdx.x;   // i indexes pairs
  if (i < n2) {
    const float2 v = ((const float2*)src)[i];
    v2h h;
    h.x = (_Float16)v.x;
    h.y = (_Float16)v.y;
    ((v2h*)dst)[i] = h;
  }
}

// ---------------------------------------------------------------------------
// WMMA GEMM:  H[M,Ncols] = Xh[M,K] @ Wh[Ncols,K]^T   (f16 operands, f32 accum)
// one wave32 per 16x16 tile; K stepped by 32 (V_WMMA_F32_16X16X32_F16).
// Operands are pre-packed f16, so each fragment is contiguous 16B/32B loads:
//   A: K in [16h, 16h+16)                      -> two v8h loads (contiguous)
//   B: K in [16h, 16h+16)                      -> two v8h loads (contiguous)
// ---------------------------------------------------------------------------
__global__ void k_gemm_f16_wmma(const _Float16* __restrict__ Xh, const _Float16* __restrict__ Wh,
                                float* __restrict__ H, int Mrows, int K, int Ncols) {
  const int wavesPerBlock = blockDim.x >> 5;
  const int wave  = blockIdx.x * wavesPerBlock + (threadIdx.x >> 5);
  const int ntile = Ncols >> 4;
  const int tiles = (Mrows >> 4) * ntile;
  if (wave >= tiles) return;
  const int tm = wave / ntile;
  const int tn = wave % ntile;

  const int lane = threadIdx.x & 31;
  const int half = lane >> 4;      // 0: lanes 0-15, 1: lanes 16-31
  const int mr   = lane & 15;

  const _Float16* __restrict__ arow = Xh + (size_t)(tm * 16 + mr) * K;  // A row M = mr
  const _Float16* __restrict__ brow = Wh + (size_t)(tn * 16 + mr) * K;  // B col N = mr

  v8f acc = {};
#pragma unroll 4
  for (int k0 = 0; k0 < K; k0 += 32) {
    const v8h a_lo = *(const v8h*)(arow + k0 + 16 * half);       // K = 16h .. 16h+7
    const v8h a_hi = *(const v8h*)(arow + k0 + 16 * half + 8);   // K = 16h+8 .. 16h+15
    const v8h b_lo = *(const v8h*)(brow + k0 + 16 * half);       // K = 16h .. 16h+7
    const v8h b_hi = *(const v8h*)(brow + k0 + 16 * half + 8);   // K = 16h+8 .. 16h+15
    const v16h a = __builtin_shufflevector(a_lo, a_hi, 0, 1, 2, 3, 4, 5, 6, 7,
                                           8, 9, 10, 11, 12, 13, 14, 15);
    const v16h b = __builtin_shufflevector(b_lo, b_hi, 0, 1, 2, 3, 4, 5, 6, 7,
                                           8, 9, 10, 11, 12, 13, 14, 15);
    acc = __builtin_amdgcn_wmma_f32_16x16x32_f16(
        /*neg_a=*/false, a, /*neg_b=*/false, b,
        /*c_mod=*/(short)0, acc, /*reuse_a=*/false, /*reuse_b=*/false);
  }

  // D 16x16 f32 layout: VGPR r -> row r+8*half, col = mr
#pragma unroll
  for (int r = 0; r < 8; ++r) {
    const int row = tm * 16 + r + 8 * half;
    const int col = tn * 16 + mr;
    H[(size_t)row * Ncols + col] = acc[r];
  }
}

// ---------------------------------------------------------------------------
// aggregation:  agg[i] = h[i]*dinv[i]^2 + bias   (self-loop term + bias init)
// ---------------------------------------------------------------------------
__global__ void k_init_agg(const float* __restrict__ h, const float* __restrict__ dinv,
                           const float* __restrict__ bias, float* __restrict__ agg,
                           int n, int C) {
  int i = blockIdx.x * blockDim.x + threadIdx.x;
  if (i >= n * C) return;
  const int node = i / C;
  const int c    = i - node * C;
  const float di = dinv[node];
  agg[i] = h[i] * (di * di) + bias[c];
}

// per-edge scatter-add: agg[dst] += h[src] * norm.  float4 gather + f32 atomics;
// the 25.6MB feature table is L2-resident (192MB), and the tid mapping keeps each
// wave on one edge at C=128 (two at C=64), so src/dst/norm loads are uniform per edge.
__global__ void k_edge_agg(const float* __restrict__ h, const long long* __restrict__ src,
                           const long long* __restrict__ dst, const float* __restrict__ nrm,
                           float* __restrict__ agg, int E, int C) {
  const int chunks = C >> 2;
  const long long tid = (long long)blockIdx.x * blockDim.x + threadIdx.x;
  if (tid >= (long long)E * chunks) return;
  const int e = (int)(tid / chunks);
  const int q = (int)(tid - (long long)e * chunks);
  const long long s = src[e];
  const long long d = dst[e];
  const float nv = nrm[e];
  const float4 hv = *(const float4*)(h + (size_t)s * C + q * 4);
  float* ap = agg + (size_t)d * C + q * 4;
  atomicAdd(ap + 0, hv.x * nv);
  atomicAdd(ap + 1, hv.y * nv);
  atomicAdd(ap + 2, hv.z * nv);
  atomicAdd(ap + 3, hv.w * nv);
}

// ---------------------------------------------------------------------------
// BatchNorm (training-mode batch stats) + ReLU, emitting f16 for the next GEMM
// ---------------------------------------------------------------------------
__global__ void k_colreduce(const float* __restrict__ a, float* __restrict__ sums,
                            float* __restrict__ sumsq, int n, int C) {
  const int c    = threadIdx.x;        // blockDim.x == C
  const int row0 = blockIdx.x * 128;
  int rend = row0 + 128;
  if (rend > n) rend = n;
  float s = 0.0f, s2 = 0.0f;
  for (int r = row0; r < rend; ++r) {
    const float v = a[(size_t)r * C + c];
    s += v;
    s2 += v * v;
  }
  atomicAdd(&sums[c], s);
  atomicAdd(&sumsq[c], s2);
}

__global__ void k_bn_relu_f16(const float* __restrict__ agg, const float* __restrict__ sums,
                              const float* __restrict__ sumsq, const float* __restrict__ gamma,
                              const float* __restrict__ beta, _Float16* __restrict__ out,
                              int n, int C) {
  int i = blockIdx.x * blockDim.x + threadIdx.x;   // i indexes f32 pairs
  if (i >= (n * C) >> 1) return;
  const float inv_n = 1.0f / (float)n;
  const float2 v2 = ((const float2*)agg)[i];
  const int base = i * 2;
  v2h o;
#pragma unroll
  for (int j = 0; j < 2; ++j) {
    const int c = (base + j) % C;
    const float mean = sums[c] * inv_n;
    const float var  = sumsq[c] * inv_n - mean * mean;  // biased variance; bias shift cancels
    const float hv   = (j == 0) ? v2.x : v2.y;
    const float hn   = (hv - mean) * rsqrtf(var + BN_EPS);
    const float v    = hn * gamma[c] + beta[c];
    const float r    = v > 0.0f ? v : 0.0f;
    if (j == 0) o.x = (_Float16)r; else o.y = (_Float16)r;
  }
  ((v2h*)out)[i] = o;
}

// ---------------------------------------------------------------------------
// launch
// ---------------------------------------------------------------------------
static inline int cdiv(long long a, long long b) { return (int)((a + b - 1) / b); }

extern "C" void kernel_launch(void* const* d_in, const int* in_sizes, int n_in,
                              void* d_out, int out_size, void* d_ws, size_t ws_size,
                              hipStream_t stream) {
  const float*     x   = (const float*)d_in[0];
  const long long* ei  = (const long long*)d_in[1];
  const float*     W1  = (const float*)d_in[2];
  const float*     b1  = (const float*)d_in[3];
  const float*     W2  = (const float*)d_in[4];
  const float*     b2  = (const float*)d_in[5];
  const float*     W3  = (const float*)d_in[6];
  const float*     b3  = (const float*)d_in[7];
  const float*     g1  = (const float*)d_in[8];
  const float*     bt1 = (const float*)d_in[9];
  const float*     g2  = (const float*)d_in[10];
  const float*     bt2 = (const float*)d_in[11];

  const int N = in_sizes[0] / IN_C;
  const int E = in_sizes[1] / 2;
  const long long* src = ei;
  const long long* dst = ei + E;

  // workspace layout:
  //   f32: dinv[N] | norm[E] | buf0[N*128] | buf1[N*128] | sums[128] | sumsq[128]
  //   f16: xh[N*128] | wh[128*128]
  float* dinv  = (float*)d_ws;
  float* nrm   = dinv + N;
  float* buf0  = nrm + E;
  float* buf1  = buf0 + (size_t)N * HID_C;
  float* sums  = buf1 + (size_t)N * HID_C;
  float* sumsq = sums + HID_C;
  _Float16* xh = (_Float16*)(sumsq + HID_C);
  _Float16* wh = xh + (size_t)N * HID_C;
  float* outp  = (float*)d_out;

  // --- GCN normalization: deg (self-loop = 1 + in-degree), dinv, per-edge norm
  k_fill<<<cdiv(N, 256), 256, 0, stream>>>(dinv, 1.0f, N);
  k_deg_accum<<<cdiv(E, 256), 256, 0, stream>>>(dst, dinv, E);
  k_rsqrt_inplace<<<cdiv(N, 256), 256, 0, stream>>>(dinv, N);
  k_norm<<<cdiv(E, 256), 256, 0, stream>>>(src, dst, dinv, nrm, E);

  const int tiles1 = (N / 16) * (HID_C / 16);
  const int tiles3 = (N / 16) * (OUT_C / 16);
  const int tot128 = N * HID_C;
  const int tot64  = N * OUT_C;

  // --- Layer 1: x->f16, W1->f16, GEMM -> buf0, agg -> buf1, BN+ReLU -> xh (f16)
  k_f32_to_f16<<<cdiv(tot128 / 2, 256), 256, 0, stream>>>(x, xh, tot128 / 2);
  k_f32_to_f16<<<cdiv((HID_C * IN_C) / 2, 256), 256, 0, stream>>>(W1, wh, (HID_C * IN_C) / 2);
  k_gemm_f16_wmma<<<cdiv(tiles1, 8), 256, 0, stream>>>(xh, wh, buf0, N, IN_C, HID_C);
  k_init_agg<<<cdiv(tot128, 256), 256, 0, stream>>>(buf0, dinv, b1, buf1, N, HID_C);
  k_edge_agg<<<cdiv((long long)E * (HID_C / 4), 256), 256, 0, stream>>>(buf0, src, dst, nrm, buf1, E, HID_C);
  k_fill<<<1, 256, 0, stream>>>(sums, 0.0f, 2 * HID_C);  // sums+sumsq contiguous
  k_colreduce<<<cdiv(N, 128), HID_C, 0, stream>>>(buf1, sums, sumsq, N, HID_C);
  k_bn_relu_f16<<<cdiv(tot128 / 2, 256), 256, 0, stream>>>(buf1, sums, sumsq, g1, bt1, xh, N, HID_C);

  // --- Layer 2
  k_f32_to_f16<<<cdiv((HID_C * HID_C) / 2, 256), 256, 0, stream>>>(W2, wh, (HID_C * HID_C) / 2);
  k_gemm_f16_wmma<<<cdiv(tiles1, 8), 256, 0, stream>>>(xh, wh, buf0, N, HID_C, HID_C);
  k_init_agg<<<cdiv(tot128, 256), 256, 0, stream>>>(buf0, dinv, b2, buf1, N, HID_C);
  k_edge_agg<<<cdiv((long long)E * (HID_C / 4), 256), 256, 0, stream>>>(buf0, src, dst, nrm, buf1, E, HID_C);
  k_fill<<<1, 256, 0, stream>>>(sums, 0.0f, 2 * HID_C);
  k_colreduce<<<cdiv(N, 128), HID_C, 0, stream>>>(buf1, sums, sumsq, N, HID_C);
  k_bn_relu_f16<<<cdiv(tot128 / 2, 256), 256, 0, stream>>>(buf1, sums, sumsq, g2, bt2, xh, N, HID_C);

  // --- Layer 3: GEMM (64ch) -> buf0, aggregate (+b3) -> d_out
  k_f32_to_f16<<<cdiv((OUT_C * HID_C) / 2, 256), 256, 0, stream>>>(W3, wh, (OUT_C * HID_C) / 2);
  k_gemm_f16_wmma<<<cdiv(tiles3, 8), 256, 0, stream>>>(xh, wh, buf0, N, HID_C, OUT_C);
  k_init_agg<<<cdiv(tot64, 256), 256, 0, stream>>>(buf0, dinv, b3, outp, N, OUT_C);
  k_edge_agg<<<cdiv((long long)E * (OUT_C / 4), 256), 256, 0, stream>>>(buf0, src, dst, nrm, outp, E, OUT_C);
}
